// CustomPatchEmbedding_17815524344571
// MI455X (gfx1250) — matrix operand order repaired and checked by reading, not compiled, not run on hardware
//
#include <hip/hip_runtime.h>
#include <hip/hip_bf16.h>

// ---------------------------------------------------------------------------
// Gathered patch-embedding as split-bf16 GEMM on CDNA5 WMMA.
//   out[b,n,e] = sum_k patch(b,n)[k] * proj_w[e,k] + proj_b[e]
//   M = 64*576 = 36864, K = 3*16*16 = 768, N = E = 768
// fp32 inputs are split into hi/lo bf16; D = Ahi*Bhi + Ahi*Blo + Alo*Bhi
// via v_wmma_f32_16x16x32_bf16 (fp32 accumulate) -> ~fp32 accuracy at ~3x
// the bf16 WMMA cost, far cheaper than the K=4 fp32 WMMA path.
// ---------------------------------------------------------------------------

typedef __attribute__((ext_vector_type(16))) __bf16 v16bf;
typedef __attribute__((ext_vector_type(8)))  __bf16 v8bf;
typedef __attribute__((ext_vector_type(8)))  float  v8f;

#define B_IMGS  64
#define C_CH    3
#define H_IMG   384
#define W_IMG   384
#define N_PATCH 576
#define E_DIM   768
#define K_DIM   768          // C*P*P
#define P_SZ    16

#define M_BLK   64           // patch rows per block
#define E_BLK   128          // output channels per block
#define K_CHUNK 32           // K per WMMA step
#define NKCHUNK (K_DIM / K_CHUNK)   // 24

union FragU { uint4 u[2]; v16bf v; };

__global__ __launch_bounds__(256)
void patch_embed_wmma_kernel(const float* __restrict__ x,
                             const int*   __restrict__ start_h,
                             const int*   __restrict__ start_w,
                             const float* __restrict__ proj_w,
                             const float* __restrict__ proj_b,
                             float*       __restrict__ out)
{
    // LDS staging: hi/lo bf16 tiles. A: [64][32], B(t): [128][32] (col-major in E)
    __shared__ __align__(16) __bf16 lds_a_hi[M_BLK * K_CHUNK];
    __shared__ __align__(16) __bf16 lds_a_lo[M_BLK * K_CHUNK];
    __shared__ __align__(16) __bf16 lds_b_hi[E_BLK * K_CHUNK];
    __shared__ __align__(16) __bf16 lds_b_lo[E_BLK * K_CHUNK];

    const int tid   = threadIdx.x;
    const int etile = blockIdx.x;            // 0..5
    const int mtile = blockIdx.y;            // 0..575
    const int m0    = mtile * M_BLK;
    const int e0    = etile * E_BLK;
    const int b     = m0 / N_PATCH;          // 576/64=9 -> block stays in one image

    // ---- A gather assignment: thread -> (row, 8 contiguous k) ----
    const int ar    = tid >> 2;              // local row 0..63
    const int akseg = tid & 3;               // which 8-wide k segment of 32
    const int am    = m0 + ar;               // global patch index == flat [b][n]
    const int ash   = start_h[am];
    const int asw   = start_w[am];

    // ---- B load assignment: thread -> (col, 16 k as 4 float4) ----
    const int bcol  = tid >> 1;              // 0..127
    const int bhalf = tid & 1;               // low/high 16 of the 32-k chunk
    const float* bsrc = proj_w + (size_t)(e0 + bcol) * K_DIM + bhalf * 16;

    float  a_reg[8];
    float4 b_reg[4];

    auto prefetch = [&](int kc) {
        // k = c*256 + ph*16 + pw ; chunk of 32 = two patch rows of one channel
        const int c   = kc >> 3;
        const int ph  = ((kc & 7) << 1) + (akseg >> 1);
        const int pw0 = (akseg & 1) << 3;
        const float* ap = x + ((size_t)(b * C_CH + c) * H_IMG + (ash + ph)) * W_IMG
                            + (asw + pw0);
        #pragma unroll
        for (int i = 0; i < 8; ++i) a_reg[i] = ap[i];    // dword-aligned gather

        const float4* bp4 = (const float4*)(bsrc + kc * K_CHUNK); // 16B aligned
        #pragma unroll
        for (int i = 0; i < 4; ++i) b_reg[i] = bp4[i];
    };

    auto stage = [&]() {
        // split fp32 -> hi/lo bf16 and stage as 16B LDS stores
        v8bf ah, al;
        #pragma unroll
        for (int i = 0; i < 8; ++i) {
            float v  = a_reg[i];
            __bf16 h = (__bf16)v;
            ah[i] = h;
            al[i] = (__bf16)(v - (float)h);
        }
        *(v8bf*)&lds_a_hi[ar * K_CHUNK + akseg * 8] = ah;
        *(v8bf*)&lds_a_lo[ar * K_CHUNK + akseg * 8] = al;

        const float* bf = (const float*)b_reg;
        v8bf bh0, bl0, bh1, bl1;
        #pragma unroll
        for (int i = 0; i < 8; ++i) {
            float v0  = bf[i];
            float v1  = bf[8 + i];
            __bf16 h0 = (__bf16)v0;
            __bf16 h1 = (__bf16)v1;
            bh0[i] = h0;  bl0[i] = (__bf16)(v0 - (float)h0);
            bh1[i] = h1;  bl1[i] = (__bf16)(v1 - (float)h1);
        }
        const int bbase = bcol * K_CHUNK + bhalf * 16;
        *(v8bf*)&lds_b_hi[bbase]     = bh0;
        *(v8bf*)&lds_b_hi[bbase + 8] = bh1;
        *(v8bf*)&lds_b_lo[bbase]     = bl0;
        *(v8bf*)&lds_b_lo[bbase + 8] = bl1;
    };

    // ---- wave tiling: 8 waves = 4(M) x 2(E); wave tile = 16 x 64 ----
    const int wave = tid >> 5;               // wave32
    const int lane = tid & 31;
    const int rw   = wave & 3;               // row group  (16 rows each)
    const int cw   = wave >> 2;              // col group  (64 cols each)
    const int lrow = lane & 15;
    const int lhi  = lane >> 4;

    // ISA 16-bit A layout: lanes0-15 VGPR0-3=K0..7, VGPR4-7=K16..23;
    //                      lanes16-31 K8..15 / K24..31.
    const int arow  = rw * 16 + lrow;
    const int aoff0 = arow * K_CHUNK + lhi * 8;
    const int aoff1 = arow * K_CHUNK + 16 + lhi * 8;

    auto load_a = [&](const __bf16* base) -> v16bf {
        FragU f;
        f.u[0] = *(const uint4*)(base + aoff0);
        f.u[1] = *(const uint4*)(base + aoff1);
        return f.v;
    };
    // ISA 16-bit B layout: lane = column; lanes0-15 hold K0..7 / K16..23,
    // lanes16-31 hold K8..15 / K24..31. LDS holds B as [col][k].
    auto load_b = [&](const __bf16* base, int col) -> v16bf {
        FragU f;
        const int o = col * K_CHUNK + lhi * 8;
        f.u[0] = *(const uint4*)(base + o);
        f.u[1] = *(const uint4*)(base + o + 16);
        return f.v;
    };

    v8f acc[4] = {v8f{}, v8f{}, v8f{}, v8f{}};

    prefetch(0);
    for (int kc = 0; kc < NKCHUNK; ++kc) {
        __syncthreads();                     // previous compute done with LDS
        stage();
        __syncthreads();                     // tiles visible
        if (kc + 1 < NKCHUNK) prefetch(kc + 1);  // overlap next gather w/ math

        const v16bf a_hi = load_a(lds_a_hi);
        const v16bf a_lo = load_a(lds_a_lo);

        #pragma unroll
        for (int j = 0; j < 4; ++j) {
            const int col = cw * 64 + j * 16 + lrow;
            const v16bf b_hi = load_b(lds_b_hi, col);
            const v16bf b_lo = load_b(lds_b_lo, col);
            acc[j] = __builtin_amdgcn_wmma_f32_16x16x32_bf16(
                         false, a_hi, false, b_hi, (short)0, acc[j], false, false);
            acc[j] = __builtin_amdgcn_wmma_f32_16x16x32_bf16(
                         false, a_hi, false, b_lo, (short)0, acc[j], false, false);
            acc[j] = __builtin_amdgcn_wmma_f32_16x16x32_bf16(
                         false, a_lo, false, b_hi, (short)0, acc[j], false, false);
        }
    }

    // ---- epilogue: C/D layout VGPR v -> rows v (lanes0-15) / v+8 (lanes16-31)
    #pragma unroll
    for (int j = 0; j < 4; ++j) {
        const int e    = e0 + cw * 64 + j * 16 + lrow;
        const float bv = proj_b[e];
        #pragma unroll
        for (int v = 0; v < 8; ++v) {
            const int m = m0 + rw * 16 + lhi * 8 + v;
            out[(size_t)m * E_DIM + e] = acc[j][v] + bv;
        }
    }
}

extern "C" void kernel_launch(void* const* d_in, const int* in_sizes, int n_in,
                              void* d_out, int out_size, void* d_ws, size_t ws_size,
                              hipStream_t stream) {
    const float* x       = (const float*)d_in[0];
    const int*   start_h = (const int*)  d_in[1];
    const int*   start_w = (const int*)  d_in[2];
    const float* proj_w  = (const float*)d_in[3];
    const float* proj_b  = (const float*)d_in[4];
    float*       out     = (float*)d_out;

    dim3 grid(E_DIM / E_BLK, (B_IMGS * N_PATCH) / M_BLK);   // (6, 576)
    patch_embed_wmma_kernel<<<grid, 256, 0, stream>>>(
        x, start_h, start_w, proj_w, proj_b, out);
}
